// MultiHeadAttention_34677565948835
// MI455X (gfx1250) — compile-verified, matrix operand order checked by reading, not run on hardware
//
#include <hip/hip_runtime.h>

#define BB   2
#define SS   4096
#define HID  1024
#define NH   16
#define BLEN 256
#define DH   64
#define NB   (SS / BLEN)      // 16
#define ATTN_SCALE 0.125f     // 64^-0.5

typedef __attribute__((ext_vector_type(16))) __bf16 v16bf;
typedef __attribute__((ext_vector_type(8)))  __bf16 v8bf;
typedef __attribute__((ext_vector_type(4)))  __bf16 v4bf;
typedef __attribute__((ext_vector_type(8)))  float  v8f;
typedef __attribute__((ext_vector_type(4)))  int    v4i;

union U16 { v16bf v; v8bf h[2]; };

#if defined(__HIP_DEVICE_COMPILE__) && defined(__gfx1250__) && \
    __has_builtin(__builtin_amdgcn_global_load_async_to_lds_b128) && \
    __has_builtin(__builtin_amdgcn_s_wait_asynccnt)
#define USE_ASYNC 1
#else
#define USE_ASYNC 0
#endif

// 16-byte global -> LDS copy: async DMA (ASYNCcnt) when available.
__device__ __forceinline__ void copy16_to_lds(const __bf16* g, __bf16* l) {
#if USE_ASYNC
  __builtin_amdgcn_global_load_async_to_lds_b128(
      (__attribute__((address_space(1))) v4i*)g,
      (__attribute__((address_space(3))) v4i*)l, 0, 0);
#else
  *(v8bf*)l = *(const v8bf*)g;
#endif
}
__device__ __forceinline__ void wait_async_copies() {
#if USE_ASYNC
  __builtin_amdgcn_s_wait_asynccnt(0);
#endif
}

__device__ __forceinline__ v8f wmma_bf16(v16bf a, v16bf b, v8f c) {
  // (neg_a, A, neg_b, B, c_mod, C, reuse_a, reuse_b)
  return __builtin_amdgcn_wmma_f32_16x16x32_bf16(false, a, false, b, (short)0, c,
                                                 false, false);
}

// ---------------------------------------------------------------------------
// Kernel 1: fp32 -> bf16 weight conversion (vectorized x4)
// ---------------------------------------------------------------------------
__global__ __launch_bounds__(256) void cvt_f32_to_bf16(const float* __restrict__ src,
                                                       __bf16* __restrict__ dst, int n4) {
  int i = blockIdx.x * blockDim.x + threadIdx.x;
  if (i < n4) {
    float4 f = ((const float4*)src)[i];
    v4bf o;
    o[0] = (__bf16)f.x; o[1] = (__bf16)f.y; o[2] = (__bf16)f.z; o[3] = (__bf16)f.w;
    ((v4bf*)dst)[i] = o;
  }
}

// ---------------------------------------------------------------------------
// Kernel 2: QKV projection GEMM.  out[m,n] = scale * sum_k X[m,k] * W[n,k]
// Double-buffered LDS, software pipelined: global loads for tile t+1 issued
// before the WMMA work of tile t; bf16 W tile copied with async-to-LDS DMA.
// vmode==0: out bf16 [B,H,S,D]; vmode==1: out bf16 [B,H,D,S] (V transposed).
// ---------------------------------------------------------------------------
__global__ __launch_bounds__(256) void gemm_qkv_kernel(
    const float* __restrict__ X, const __bf16* __restrict__ W,
    __bf16* __restrict__ out, float scale, int vmode) {
  __shared__ __align__(16) __bf16 As[2][128][40];
  __shared__ __align__(16) __bf16 Bs[2][128][40];
  const int m0 = blockIdx.x * 128;
  const int n0 = blockIdx.y * 128;
  const int tid  = threadIdx.x;
  const int lane = tid & 31;
  const int wave = tid >> 5;
  const int wm = wave & 3;           // 4 row-waves
  const int wn = wave >> 2;          // 2 col-waves
  const int laneM = lane & 15;
  const int koff  = (lane < 16) ? 0 : 8;
  const int rhalf = (lane < 16) ? 0 : 8;
  const int rA = tid >> 3, cA = (tid & 7) << 2;   // A: 32 rows x 8 float4 cols
  const int rB = tid >> 2, cB = (tid & 3) << 3;   // B: 64 rows x 4 v8bf cols

  const float*  xA = X + (size_t)(m0 + rA) * HID + cA;
  const __bf16* wB = W + (size_t)(n0 + rB) * HID + cB;

  v8f acc[2][4] = {};
  float4 fa[4];

  // ---- prologue: stage k-tile 0 into buffer 0 ----
#pragma unroll
  for (int it = 0; it < 4; ++it)
    fa[it] = *(const float4*)(xA + (size_t)it * 32 * HID);
#pragma unroll
  for (int it = 0; it < 2; ++it)
    copy16_to_lds(wB + (size_t)it * 64 * HID, &Bs[0][rB + it * 64][cB]);
#pragma unroll
  for (int it = 0; it < 4; ++it) {
    v4bf o;
    o[0] = (__bf16)fa[it].x; o[1] = (__bf16)fa[it].y;
    o[2] = (__bf16)fa[it].z; o[3] = (__bf16)fa[it].w;
    *(v4bf*)&As[0][rA + it * 32][cA] = o;
  }
  wait_async_copies();
  __syncthreads();

  for (int kt = 0; kt < HID / 32; ++kt) {
    const int cur = kt & 1, nxt = cur ^ 1;
    const bool more = (kt + 1) < (HID / 32);
    const int k1 = (kt + 1) * 32;

    if (more) {   // issue next tile's global traffic before computing
#pragma unroll
      for (int it = 0; it < 4; ++it)
        fa[it] = *(const float4*)(xA + (size_t)it * 32 * HID + k1);
#pragma unroll
      for (int it = 0; it < 2; ++it)
        copy16_to_lds(wB + (size_t)it * 64 * HID + k1, &Bs[nxt][rB + it * 64][cB]);
    }

    U16 af[2];
#pragma unroll
    for (int mi = 0; mi < 2; ++mi) {
      const __bf16* p = &As[cur][wm * 32 + mi * 16 + laneM][koff];
      af[mi].h[0] = *(const v8bf*)p;
      af[mi].h[1] = *(const v8bf*)(p + 16);
    }
    U16 bfr[4];
#pragma unroll
    for (int ni = 0; ni < 4; ++ni) {
      const __bf16* p = &Bs[cur][wn * 64 + ni * 16 + laneM][koff];
      bfr[ni].h[0] = *(const v8bf*)p;
      bfr[ni].h[1] = *(const v8bf*)(p + 16);
    }
#pragma unroll
    for (int mi = 0; mi < 2; ++mi)
#pragma unroll
      for (int ni = 0; ni < 4; ++ni)
        acc[mi][ni] = wmma_bf16(af[mi].v, bfr[ni].v, acc[mi][ni]);

    if (more) {   // convert+store A after compute (loads had time to land)
#pragma unroll
      for (int it = 0; it < 4; ++it) {
        v4bf o;
        o[0] = (__bf16)fa[it].x; o[1] = (__bf16)fa[it].y;
        o[2] = (__bf16)fa[it].z; o[3] = (__bf16)fa[it].w;
        *(v4bf*)&As[nxt][rA + it * 32][cA] = o;
      }
      wait_async_copies();
    }
    __syncthreads();
  }

#pragma unroll
  for (int mi = 0; mi < 2; ++mi)
#pragma unroll
    for (int ni = 0; ni < 4; ++ni)
#pragma unroll
      for (int r = 0; r < 8; ++r) {
        int m = m0 + wm * 32 + mi * 16 + r + rhalf;
        int n = n0 + wn * 64 + ni * 16 + laneM;
        int b = m >> 12, s = m & (SS - 1);
        int h = n >> 6,  d = n & (DH - 1);
        float val = acc[mi][ni][r] * scale;
        size_t off = vmode ? ((((size_t)b * NH + h) * DH + d) * SS + s)
                           : ((((size_t)b * NH + h) * SS + s) * DH + d);
        out[off] = (__bf16)val;
      }
}

// ---------------------------------------------------------------------------
// Kernel 3: flash-style windowed causal attention.
// Block idx -> (b, h, seq-block). 8 waves, 32 query rows each.
// Window: block0 -> [0,256), block n -> [(n-1)*256, (n+1)*256); mask sk > sq.
// ---------------------------------------------------------------------------
__global__ __launch_bounds__(256) void attn_kernel(
    const __bf16* __restrict__ qh, const __bf16* __restrict__ kh,
    const __bf16* __restrict__ vt, __bf16* __restrict__ attn_out) {
  __shared__ __align__(16) __bf16 Pw[8][32][72];   // per-wave P staging
  const int idx = blockIdx.x;
  const int blk = idx & (NB - 1);
  const int bh  = idx >> 4;
  const int b   = bh >> 4;
  const int h   = bh & (NH - 1);
  const int lane = threadIdx.x & 31;
  const int wave = threadIdx.x >> 5;
  const int laneM = lane & 15;
  const int koff  = (lane < 16) ? 0 : 8;
  const int rhalf = (lane < 16) ? 0 : 8;

  const int qrow0  = blk * BLEN + wave * 32;
  const int kstart = (blk == 0) ? 0 : (blk - 1) * BLEN;
  const int nch    = (blk == 0) ? (BLEN / 64) : (2 * BLEN / 64);

  const __bf16* qb = qh + (size_t)bh * SS * DH;
  const __bf16* kb = kh + (size_t)bh * SS * DH;
  const __bf16* vb = vt + (size_t)bh * DH * SS;

  U16 qa[2][2];                       // Q fragments, resident for whole block
#pragma unroll
  for (int mi = 0; mi < 2; ++mi)
#pragma unroll
    for (int ks = 0; ks < 2; ++ks) {
      const __bf16* p = qb + (size_t)(qrow0 + mi * 16 + laneM) * DH + ks * 32 + koff;
      qa[mi][ks].h[0] = *(const v8bf*)p;
      qa[mi][ks].h[1] = *(const v8bf*)(p + 16);
    }

  float mrow[16], srow[16];
#pragma unroll
  for (int i = 0; i < 16; ++i) { mrow[i] = -1e30f; srow[i] = 0.0f; }
  v8f o[2][4] = {};

  for (int c = 0; c < nch; ++c) {
    const int sk0 = kstart + c * 64;
    if (sk0 > qrow0 + 31) break;      // chunk fully above the diagonal

    // ---- S = Q * K^T (32 x 64 chunk), fp32 accum ----
    v8f sf[2][4];
#pragma unroll
    for (int ni = 0; ni < 4; ++ni) {
      U16 kf0, kf1;
      const __bf16* p = kb + (size_t)(sk0 + ni * 16 + laneM) * DH + koff;
      kf0.h[0] = *(const v8bf*)p;        kf0.h[1] = *(const v8bf*)(p + 16);
      kf1.h[0] = *(const v8bf*)(p + 32); kf1.h[1] = *(const v8bf*)(p + 48);
#pragma unroll
      for (int mi = 0; mi < 2; ++mi) {
        v8f z = {};
        z = wmma_bf16(qa[mi][0].v, kf0.v, z);
        sf[mi][ni] = wmma_bf16(qa[mi][1].v, kf1.v, z);
      }
    }

    // ---- causal mask + chunk row-max ----
    float cmax[16];
#pragma unroll
    for (int i = 0; i < 16; ++i) cmax[i] = -1e30f;
#pragma unroll
    for (int mi = 0; mi < 2; ++mi)
#pragma unroll
      for (int ni = 0; ni < 4; ++ni)
#pragma unroll
        for (int r = 0; r < 8; ++r) {
          int sq = qrow0 + mi * 16 + r + rhalf;
          int sk = sk0 + ni * 16 + laneM;
          float vv = sf[mi][ni][r];
          if (sk > sq) vv = -1e9f;
          sf[mi][ni][r] = vv;
          cmax[mi * 8 + r] = fmaxf(cmax[mi * 8 + r], vv);
        }
#pragma unroll
    for (int i = 0; i < 16; ++i) {
      cmax[i] = fmaxf(cmax[i], __shfl_xor(cmax[i], 1, 32));
      cmax[i] = fmaxf(cmax[i], __shfl_xor(cmax[i], 2, 32));
      cmax[i] = fmaxf(cmax[i], __shfl_xor(cmax[i], 4, 32));
      cmax[i] = fmaxf(cmax[i], __shfl_xor(cmax[i], 8, 32));
    }

    // ---- online softmax rescale ----
    float alpha[16];
#pragma unroll
    for (int i = 0; i < 16; ++i) {
      float mn = fmaxf(mrow[i], cmax[i]);
      alpha[i] = __expf(mrow[i] - mn);
      mrow[i]  = mn;
      srow[i] *= alpha[i];
    }
#pragma unroll
    for (int mi = 0; mi < 2; ++mi)
#pragma unroll
      for (int dt = 0; dt < 4; ++dt)
#pragma unroll
        for (int r = 0; r < 8; ++r) o[mi][dt][r] *= alpha[mi * 8 + r];

    // ---- P = exp(S - m), row sums, stage P to LDS (C-layout -> A-layout) ----
    float csum[16];
#pragma unroll
    for (int i = 0; i < 16; ++i) csum[i] = 0.0f;
#pragma unroll
    for (int mi = 0; mi < 2; ++mi)
#pragma unroll
      for (int ni = 0; ni < 4; ++ni)
#pragma unroll
        for (int r = 0; r < 8; ++r) {
          float pe = __expf(sf[mi][ni][r] - mrow[mi * 8 + r]);
          csum[mi * 8 + r] += pe;
          Pw[wave][mi * 16 + r + rhalf][ni * 16 + laneM] = (__bf16)pe;
        }
#pragma unroll
    for (int i = 0; i < 16; ++i) {
      csum[i] += __shfl_xor(csum[i], 1, 32);
      csum[i] += __shfl_xor(csum[i], 2, 32);
      csum[i] += __shfl_xor(csum[i], 4, 32);
      csum[i] += __shfl_xor(csum[i], 8, 32);
      srow[i] += csum[i];
    }

    asm volatile("s_wait_dscnt 0" ::: "memory");   // wave-private LDS ordering

    // ---- O += P * V (V pre-transposed [D,S] -> contiguous B-frag loads) ----
    U16 pa[2][2];
#pragma unroll
    for (int mi = 0; mi < 2; ++mi)
#pragma unroll
      for (int ks = 0; ks < 2; ++ks) {
        const __bf16* p = &Pw[wave][mi * 16 + laneM][ks * 32 + koff];
        pa[mi][ks].h[0] = *(const v8bf*)p;
        pa[mi][ks].h[1] = *(const v8bf*)(p + 16);
      }
#pragma unroll
    for (int dt = 0; dt < 4; ++dt) {
      U16 vf0, vf1;
      const __bf16* p = vb + (size_t)(dt * 16 + laneM) * SS + sk0 + koff;
      vf0.h[0] = *(const v8bf*)p;        vf0.h[1] = *(const v8bf*)(p + 16);
      vf1.h[0] = *(const v8bf*)(p + 32); vf1.h[1] = *(const v8bf*)(p + 48);
#pragma unroll
      for (int mi = 0; mi < 2; ++mi) {
        o[mi][dt] = wmma_bf16(pa[mi][0].v, vf0.v, o[mi][dt]);
        o[mi][dt] = wmma_bf16(pa[mi][1].v, vf1.v, o[mi][dt]);
      }
    }
  }

#pragma unroll
  for (int i = 0; i < 16; ++i) srow[i] = 1.0f / srow[i];
#pragma unroll
  for (int mi = 0; mi < 2; ++mi)
#pragma unroll
    for (int dt = 0; dt < 4; ++dt)
#pragma unroll
      for (int r = 0; r < 8; ++r) {
        int sq  = qrow0 + mi * 16 + r + rhalf;
        int col = h * DH + dt * 16 + laneM;
        attn_out[((size_t)b * SS + sq) * HID + col] =
            (__bf16)(o[mi][dt][r] * srow[mi * 8 + r]);
      }
}

// ---------------------------------------------------------------------------
// Kernel 4: output projection. out[m,n] = sum_k A[m,k] * Wo[n,k], fp32 out.
// Both tiles bf16 -> both staged with async-to-LDS DMA; double buffered.
// ---------------------------------------------------------------------------
__global__ __launch_bounds__(256) void gemm_out_kernel(
    const __bf16* __restrict__ A, const __bf16* __restrict__ W,
    float* __restrict__ out) {
  __shared__ __align__(16) __bf16 As[2][128][40];
  __shared__ __align__(16) __bf16 Bs[2][128][40];
  const int m0 = blockIdx.x * 128;
  const int n0 = blockIdx.y * 128;
  const int tid  = threadIdx.x;
  const int lane = tid & 31;
  const int wave = tid >> 5;
  const int wm = wave & 3;
  const int wn = wave >> 2;
  const int laneM = lane & 15;
  const int koff  = (lane < 16) ? 0 : 8;
  const int rhalf = (lane < 16) ? 0 : 8;
  const int rB = tid >> 2, cB = (tid & 3) << 3;

  const __bf16* aA = A + (size_t)(m0 + rB) * HID + cB;
  const __bf16* wB = W + (size_t)(n0 + rB) * HID + cB;

  v8f acc[2][4] = {};

#pragma unroll
  for (int it = 0; it < 2; ++it) {
    copy16_to_lds(aA + (size_t)it * 64 * HID, &As[0][rB + it * 64][cB]);
    copy16_to_lds(wB + (size_t)it * 64 * HID, &Bs[0][rB + it * 64][cB]);
  }
  wait_async_copies();
  __syncthreads();

  for (int kt = 0; kt < HID / 32; ++kt) {
    const int cur = kt & 1, nxt = cur ^ 1;
    const bool more = (kt + 1) < (HID / 32);
    const int k1 = (kt + 1) * 32;

    if (more) {
#pragma unroll
      for (int it = 0; it < 2; ++it) {
        copy16_to_lds(aA + (size_t)it * 64 * HID + k1, &As[nxt][rB + it * 64][cB]);
        copy16_to_lds(wB + (size_t)it * 64 * HID + k1, &Bs[nxt][rB + it * 64][cB]);
      }
    }

    U16 af[2];
#pragma unroll
    for (int mi = 0; mi < 2; ++mi) {
      const __bf16* p = &As[cur][wm * 32 + mi * 16 + laneM][koff];
      af[mi].h[0] = *(const v8bf*)p;
      af[mi].h[1] = *(const v8bf*)(p + 16);
    }
    U16 bfr[4];
#pragma unroll
    for (int ni = 0; ni < 4; ++ni) {
      const __bf16* p = &Bs[cur][wn * 64 + ni * 16 + laneM][koff];
      bfr[ni].h[0] = *(const v8bf*)p;
      bfr[ni].h[1] = *(const v8bf*)(p + 16);
    }
#pragma unroll
    for (int mi = 0; mi < 2; ++mi)
#pragma unroll
      for (int ni = 0; ni < 4; ++ni)
        acc[mi][ni] = wmma_bf16(af[mi].v, bfr[ni].v, acc[mi][ni]);

    if (more) wait_async_copies();
    __syncthreads();
  }

#pragma unroll
  for (int mi = 0; mi < 2; ++mi)
#pragma unroll
    for (int ni = 0; ni < 4; ++ni)
#pragma unroll
      for (int r = 0; r < 8; ++r) {
        int m = m0 + wm * 32 + mi * 16 + r + rhalf;
        int n = n0 + wn * 64 + ni * 16 + laneM;
        out[(size_t)m * HID + n] = acc[mi][ni][r];
      }
}

// ---------------------------------------------------------------------------
extern "C" void kernel_launch(void* const* d_in, const int* in_sizes, int n_in,
                              void* d_out, int out_size, void* d_ws, size_t ws_size,
                              hipStream_t stream) {
  (void)in_sizes; (void)n_in; (void)out_size; (void)ws_size;
  const float* q  = (const float*)d_in[0];
  const float* k  = (const float*)d_in[1];
  const float* v  = (const float*)d_in[2];
  // d_in[3] = mask (bool) — computed analytically on device, unused
  const float* Wq = (const float*)d_in[4];
  const float* Wk = (const float*)d_in[5];
  const float* Wv = (const float*)d_in[6];
  const float* Wo = (const float*)d_in[7];

  const size_t WBYTES = (size_t)HID * HID * 2;        // 2 MB per weight (bf16)
  const size_t ABYTES = (size_t)BB * SS * HID * 2;    // 16 MB per activation (bf16)
  char* wsb = (char*)d_ws;
  __bf16* wqb = (__bf16*)(wsb + 0 * WBYTES);
  __bf16* wkb = (__bf16*)(wsb + 1 * WBYTES);
  __bf16* wvb = (__bf16*)(wsb + 2 * WBYTES);
  __bf16* wob = (__bf16*)(wsb + 3 * WBYTES);
  __bf16* qhb = (__bf16*)(wsb + 4 * WBYTES);               // [B,H,S,D]
  __bf16* khb = (__bf16*)(wsb + 4 * WBYTES + 1 * ABYTES);  // [B,H,S,D]
  __bf16* vtb = (__bf16*)(wsb + 4 * WBYTES + 2 * ABYTES);  // [B,H,D,S]
  __bf16* atb = (__bf16*)(wsb + 4 * WBYTES + 3 * ABYTES);  // [B,S,HID]

  const int n4 = (HID * HID) / 4;
  cvt_f32_to_bf16<<<dim3(n4 / 256), dim3(256), 0, stream>>>(Wq, wqb, n4);
  cvt_f32_to_bf16<<<dim3(n4 / 256), dim3(256), 0, stream>>>(Wk, wkb, n4);
  cvt_f32_to_bf16<<<dim3(n4 / 256), dim3(256), 0, stream>>>(Wv, wvb, n4);
  cvt_f32_to_bf16<<<dim3(n4 / 256), dim3(256), 0, stream>>>(Wo, wob, n4);

  dim3 gg(BB * SS / 128, HID / 128), bb(256);
  gemm_qkv_kernel<<<gg, bb, 0, stream>>>(q, wqb, qhb, ATTN_SCALE, 0);
  gemm_qkv_kernel<<<gg, bb, 0, stream>>>(k, wkb, khb, 1.0f, 0);
  gemm_qkv_kernel<<<gg, bb, 0, stream>>>(v, wvb, vtb, 1.0f, 1);

  attn_kernel<<<dim3(BB * NH * NB), bb, 0, stream>>>(qhb, khb, vtb, atb);

  gemm_out_kernel<<<gg, bb, 0, stream>>>(atb, wob, (float*)d_out);
}
